// WindowAttention3D_82703890252431
// MI455X (gfx1250) — matrix operand order repaired and checked by reading, not compiled, not run on hardware
//
#include <hip/hip_runtime.h>
#include <hip/hip_bf16.h>

typedef __attribute__((ext_vector_type(16))) _Float16 v16h;
typedef __attribute__((ext_vector_type(8)))  _Float16 v8h;
typedef __attribute__((ext_vector_type(8)))  float    v8f;

// gfx1250 async global->LDS builtins (ASYNCcnt path); fallback is synchronous
// global_load_b128 + ds_store_b128 staging.
#if defined(__has_builtin)
#if __has_builtin(__builtin_amdgcn_global_load_async_to_lds_b128) && \
    __has_builtin(__builtin_amdgcn_s_wait_asynccnt)
#define HAVE_ASYNC_LDS 1
#endif
#endif

// Builtin signature (from hipcc diagnostic): param 0 = AS1 int4*, param 1 = AS3 int4*.
typedef int v4i_gcc __attribute__((vector_size(16)));
typedef __attribute__((address_space(1))) v4i_gcc gv4i;
typedef __attribute__((address_space(3))) v4i_gcc sv4i;

#ifdef HAVE_ASYNC_LDS
#define STAGE_WAIT(n) __builtin_amdgcn_s_wait_asynccnt(n)
#else
#define STAGE_WAIT(n) ((void)0)
#endif

// ---------------------------------------------------------------------------
// WMMA helpers (gfx1250: V_WMMA_F32_16X16X32_F16, wave32)
// ---------------------------------------------------------------------------
__device__ __forceinline__ v8f wmma32(v16h a, v16h b, v8f c) {
    // (neg_a, A, neg_b, B, c_mod, C, reuse_a, reuse_b)
    return __builtin_amdgcn_wmma_f32_16x16x32_f16(false, a, false, b, (short)0, c,
                                                  false, false);
}

// A fragment (16x32, M x K), row-major source with given element stride.
// Lane L: row = L&15, half = L>>4. Element j<8 -> k = half*8 + j,
// j>=8 -> k = 16 + half*8 + (j-8).  -> two contiguous 8-half (16B) chunks.
__device__ __forceinline__ v16h frag_a(const _Float16* base, int stride, int lane) {
    int row  = lane & 15;
    int half = lane >> 4;
    const _Float16* rp = base + row * stride;
    v8h c0 = *(const v8h*)(rp + half * 8);
    v8h c1 = *(const v8h*)(rp + 16 + half * 8);
    v16h r;
    #pragma unroll
    for (int i = 0; i < 8; ++i) { r[i] = c0[i]; r[8 + i] = c1[i]; }
    return r;
}

// B fragment (32x16, K x N). Source holds Bmat[n][k] row-major (i.e. the
// "column" n is a contiguous row of K values) with given stride.
// Lane L: n = L&15, j<8 -> k = (L>>4)*8 + j, j>=8 -> k = 16 + (L>>4)*8 + (j-8).
__device__ __forceinline__ v16h frag_b(const _Float16* base, int stride, int lane) {
    int col = lane & 15;
    int kb  = (lane >> 4) * 8;
    const _Float16* rp = base + col * stride + kb;
    v8h c0 = *(const v8h*)(rp);
    v8h c1 = *(const v8h*)(rp + 16);
    v16h r;
    #pragma unroll
    for (int i = 0; i < 8; ++i) { r[i] = c0[i]; r[8 + i] = c1[i]; }
    return r;
}

// Stage 32 bytes (16 halves) global -> LDS (2 async b128 ops per call).
__device__ __forceinline__ void stage32(const _Float16* gsrc, _Float16* ldst) {
#ifdef HAVE_ASYNC_LDS
    gv4i* g = (gv4i*)(unsigned long long)gsrc;
    sv4i* s = (sv4i*)(unsigned int)(unsigned long long)ldst;  // low 32b = LDS offset
    __builtin_amdgcn_global_load_async_to_lds_b128(g, s, 0, 0);
    __builtin_amdgcn_global_load_async_to_lds_b128(g, s, 16, 0);
#else
    const uint4* src = (const uint4*)gsrc;
    uint4 v0 = src[0], v1 = src[1];
    uint4* dst = (uint4*)ldst;
    dst[0] = v0; dst[1] = v1;
#endif
}

// ---------------------------------------------------------------------------
// Constants
// ---------------------------------------------------------------------------
#define NTOK   262144      // 4096 windows * 64 tokens
#define CDIM   256
#define NWIN   4096
#define NSPAT  262144      // 64*64*64

// ---------------------------------------------------------------------------
// Kernel 1: pack x (1,C,64,64,64) fp32 -> tok[t][c] fp16, window-partitioned
// ---------------------------------------------------------------------------
__global__ void pack_x_kernel(const float* __restrict__ x, _Float16* __restrict__ tok) {
    int idx = blockIdx.x * 256 + threadIdx.x;     // 0 .. C*NSPAT-1
    int c  = idx >> 18;                           // / 262144
    int sp = idx & (NSPAT - 1);
    int D = sp >> 12, H = (sp >> 6) & 63, W = sp & 63;
    int win = ((D >> 2) * 16 + (H >> 2)) * 16 + (W >> 2);
    int s   = ((D & 3) * 4 + (H & 3)) * 4 + (W & 3);
    int t   = win * 64 + s;
    tok[(size_t)t * CDIM + c] = (_Float16)x[idx];
}

__global__ void cvt_f16_kernel(const float* __restrict__ src, _Float16* __restrict__ dst, int n) {
    int i = blockIdx.x * 256 + threadIdx.x;
    if (i < n) dst[i] = (_Float16)src[i];
}

// ---------------------------------------------------------------------------
// GEMM: Out[t, m] = sum_k A[t,k] * W[m,k] + bias[m]   (K = 256 fixed)
// Block: 256 threads (8 waves). Tile: 128 tokens x 128 out-channels.
// Wave (wm = wv&3, wn = wv>>2) computes 32x64 = 2x4 WMMA tiles.
// Double-buffered LDS: tile i+1 streams in via ASYNCcnt while tile i is
// consumed; s_wait_asynccnt 4 retires exactly the current tile (in-order,
// 4 async ops per thread per tile).
// mode 0: scatter to q/k/vt layouts (QKV projection, q scaled by 1/8)
// mode 1: plain fp16 [t][m] store
// mode 2: bias + fp32 scatter back to NCDHW with window reverse
// ---------------------------------------------------------------------------
__global__ void __launch_bounds__(256)
gemm_wmma_kernel(const _Float16* __restrict__ A, const _Float16* __restrict__ W,
                 const float* __restrict__ bias, int mode,
                 _Float16* __restrict__ oq, _Float16* __restrict__ ok,
                 _Float16* __restrict__ ovt, _Float16* __restrict__ oh,
                 float* __restrict__ of) {
    __shared__ _Float16 As[2][128 * 40];   // 128 rows x 32 halves (+8 pad), x2 buffers
    __shared__ _Float16 Bs[2][128 * 40];

    const int tid  = threadIdx.x;
    const int lane = tid & 31;
    const int wv   = tid >> 5;
    const int wm   = wv & 3;            // 0..3  (row block of 32)
    const int wn   = wv >> 2;           // 0..1  (col block of 64)
    const int rowBase = blockIdx.x * 128;
    const int colBase = blockIdx.y * 128;

    const int sr   = tid >> 1;          // staging row 0..127
    const int soff = (tid & 1) * 16;    // staging chunk offset (halves)
    const _Float16* gA = A + (size_t)(rowBase + sr) * 256 + soff;
    const _Float16* gW = W + (size_t)(colBase + sr) * 256 + soff;
    const int sdst = sr * 40 + soff;

    v8f acc[2][4];
    #pragma unroll
    for (int i = 0; i < 2; ++i)
        #pragma unroll
        for (int j = 0; j < 4; ++j) acc[i][j] = {};

    // prologue: stage tile 0 into buffer 0 (4 async ops per thread)
    stage32(gA, &As[0][sdst]);
    stage32(gW, &Bs[0][sdst]);

    for (int it = 0; it < 8; ++it) {
        const int cur = it & 1;
        if (it < 7) {
            // stream next K-slab into the other buffer, then retire current
            int kn = (it + 1) * 32;
            stage32(gA + kn, &As[cur ^ 1][sdst]);
            stage32(gW + kn, &Bs[cur ^ 1][sdst]);
            STAGE_WAIT(4);
        } else {
            STAGE_WAIT(0);
        }
        __syncthreads();   // current buffer fully populated for all waves

        v16h af[2], bf[4];
        #pragma unroll
        for (int tm = 0; tm < 2; ++tm)
            af[tm] = frag_a(&As[cur][(wm * 32 + tm * 16) * 40], 40, lane);
        #pragma unroll
        for (int tn = 0; tn < 4; ++tn)
            bf[tn] = frag_b(&Bs[cur][(wn * 64 + tn * 16) * 40], 40, lane);

        #pragma unroll
        for (int tm = 0; tm < 2; ++tm)
            #pragma unroll
            for (int tn = 0; tn < 4; ++tn)
                acc[tm][tn] = wmma32(af[tm], bf[tn], acc[tm][tn]);
        __syncthreads();   // all waves done reading 'cur' before it is restaged
    }

    // ---- epilogue: D tile layout: lane -> n = lane&15 (out channel),
    //               element r -> m = r + (lane>>4)*8 (token)
    #pragma unroll
    for (int tm = 0; tm < 2; ++tm) {
        #pragma unroll
        for (int tn = 0; tn < 4; ++tn) {
            int col   = colBase + wn * 64 + tn * 16 + (lane & 15);
            int rbase = rowBase + wm * 32 + tm * 16 + (lane >> 4) * 8;
            float bv = bias[col];
            if (mode == 0) {
                int d = col & 63, hh = (col >> 6) & 3, which = col >> 8;
                #pragma unroll
                for (int r = 0; r < 8; ++r) {
                    float v = acc[tm][tn][r] + bv;
                    int t = rbase + r, win = t >> 6, s = t & 63;
                    size_t hb = (size_t)(win * 4 + hh) * 4096;  // 64*64 per (win,head)
                    if (which == 0)      oq [hb + s * 64 + d] = (_Float16)(v * 0.125f);
                    else if (which == 1) ok [hb + s * 64 + d] = (_Float16)v;
                    else                 ovt[hb + d * 64 + s] = (_Float16)v;
                }
            } else if (mode == 1) {
                #pragma unroll
                for (int r = 0; r < 8; ++r)
                    oh[(size_t)(rbase + r) * 256 + col] = (_Float16)(acc[tm][tn][r] + bv);
            } else {
                #pragma unroll
                for (int r = 0; r < 8; ++r) {
                    int t = rbase + r, win = t >> 6, s = t & 63;
                    int nw = win & 15, nh = (win >> 4) & 15, nd = win >> 8;
                    int ww = s & 3, wh2 = (s >> 2) & 3, wd = s >> 4;
                    int D = nd * 4 + wd, H = nh * 4 + wh2, Wc = nw * 4 + ww;
                    of[(size_t)col * NSPAT + D * 4096 + H * 64 + Wc] =
                        acc[tm][tn][r] + bv;
                }
            }
        }
    }
}

// ---------------------------------------------------------------------------
// Attention: one block per (window, head). 128 threads = 4 waves.
// Wave w owns 16 query rows: scores (QK^T) -> softmax -> P*V, all WMMA.
// q/k: [win][h][s][d] row-major, vt: [win][h][d][s] row-major, all fp16.
// Output written into o[t][c] fp16 (c = h*64 + d).
// ---------------------------------------------------------------------------
__global__ void __launch_bounds__(128)
attn_win_kernel(const _Float16* __restrict__ q, const _Float16* __restrict__ k,
                const _Float16* __restrict__ vt, _Float16* __restrict__ o) {
    __shared__ float    sc[4][16][68];   // per-wave score rows (padded)
    __shared__ _Float16 pb[4][16][72];   // per-wave prob rows (padded)

    const int tid  = threadIdx.x;
    const int lane = tid & 31;
    const int w    = tid >> 5;
    const int win  = blockIdx.x >> 2;
    const int h    = blockIdx.x & 3;
    const size_t hb = (size_t)(win * 4 + h) * 4096;
    const _Float16* qb = q  + hb + (size_t)w * 16 * 64;   // this wave's 16 rows
    const _Float16* kb = k  + hb;
    const _Float16* vb = vt + hb;

    // warm V tile into cache while QK^T runs (global_prefetch_b8)
    __builtin_prefetch(vb + (size_t)tid * 32, 0, 1);

    // ---- scores: S[16 x 64] for this wave, K-dim = head dim (64)
    #pragma unroll
    for (int nt = 0; nt < 4; ++nt) {
        v8f accs = {};
        #pragma unroll
        for (int kk = 0; kk < 64; kk += 32) {
            v16h a = frag_a(qb + kk, 64, lane);
            v16h b = frag_b(kb + nt * 16 * 64 + kk, 64, lane);
            accs = wmma32(a, b, accs);
        }
        int coln = nt * 16 + (lane & 15);
        int rb   = (lane >> 4) * 8;
        #pragma unroll
        for (int r = 0; r < 8; ++r) sc[w][rb + r][coln] = accs[r];
    }
    __syncthreads();

    // ---- softmax: lanes 0..15 of each wave each own one full row
    if (lane < 16) {
        float mx = -3.0e38f;
        for (int i = 0; i < 64; ++i) mx = fmaxf(mx, sc[w][lane][i]);
        float sum = 0.f;
        for (int i = 0; i < 64; ++i) sum += __expf(sc[w][lane][i] - mx);
        float inv = 1.0f / sum;
        for (int i = 0; i < 64; ++i)
            pb[w][lane][i] = (_Float16)(__expf(sc[w][lane][i] - mx) * inv);
    }
    __syncthreads();

    // ---- O = P @ V : K-dim = 64 keys, N = 64 head dims (vt is [d][s])
    const _Float16* pbase = &pb[w][0][0];
    #pragma unroll
    for (int nt = 0; nt < 4; ++nt) {
        v8f acco = {};
        #pragma unroll
        for (int kk = 0; kk < 64; kk += 32) {
            v16h a = frag_a(pbase + kk, 72, lane);
            v16h b = frag_b(vb + nt * 16 * 64 + kk, 64, lane);
            acco = wmma32(a, b, acco);
        }
        int dout = nt * 16 + (lane & 15);
        int c    = h * 64 + dout;
        int t0   = win * 64 + w * 16 + (lane >> 4) * 8;
        #pragma unroll
        for (int r = 0; r < 8; ++r)
            o[(size_t)(t0 + r) * 256 + c] = (_Float16)acco[r];
    }
}

// ---------------------------------------------------------------------------
// Host launcher
// ---------------------------------------------------------------------------
extern "C" void kernel_launch(void* const* d_in, const int* in_sizes, int n_in,
                              void* d_out, int out_size, void* d_ws, size_t ws_size,
                              hipStream_t stream) {
    const float* x    = (const float*)d_in[0];
    const float* inw  = (const float*)d_in[1];  // (768,256)
    const float* inb  = (const float*)d_in[2];  // (768,)
    const float* outw = (const float*)d_in[3];  // (256,256)
    const float* outb = (const float*)d_in[4];
    const float* prw  = (const float*)d_in[5];  // (256,256)
    const float* prb  = (const float*)d_in[6];
    float* out = (float*)d_out;

    // workspace carve (all sizes multiples of 256B)
    char* p = (char*)d_ws;
    const size_t SZ_TC = (size_t)NTOK * CDIM * 2;   // 134 MB fp16 [t][c]
    _Float16* tok = (_Float16*)p; p += SZ_TC;       // tokens, later reused for attn out
    _Float16* q   = (_Float16*)p; p += SZ_TC;
    _Float16* kk  = (_Float16*)p; p += SZ_TC;
    _Float16* vt  = (_Float16*)p; p += SZ_TC;
    _Float16* y1  = (_Float16*)p; p += SZ_TC;
    _Float16* wq  = (_Float16*)p; p += (size_t)768 * 256 * 2;
    _Float16* wo  = (_Float16*)p; p += (size_t)256 * 256 * 2;
    _Float16* wp  = (_Float16*)p; p += (size_t)256 * 256 * 2;

    // 1) pack + weight conversion
    pack_x_kernel<<<NTOK * CDIM / 256, 256, 0, stream>>>(x, tok);
    cvt_f16_kernel<<<(768 * 256 + 255) / 256, 256, 0, stream>>>(inw, wq, 768 * 256);
    cvt_f16_kernel<<<(256 * 256 + 255) / 256, 256, 0, stream>>>(outw, wo, 256 * 256);
    cvt_f16_kernel<<<(256 * 256 + 255) / 256, 256, 0, stream>>>(prw, wp, 256 * 256);

    // 2) QKV projection: [262144 x 256] x [768 x 256]^T, scatter to q/k/vt
    gemm_wmma_kernel<<<dim3(NTOK / 128, 768 / 128), 256, 0, stream>>>(
        tok, wq, inb, 0, q, kk, vt, nullptr, nullptr);

    // 3) windowed attention (4096 windows x 4 heads); writes into tok buffer
    attn_win_kernel<<<NWIN * 4, 128, 0, stream>>>(q, kk, vt, tok);

    // 4) out_proj: [262144 x 256] x [256 x 256]^T -> y1 (fp16)
    gemm_wmma_kernel<<<dim3(NTOK / 128, 256 / 128), 256, 0, stream>>>(
        tok, wo, outb, 1, nullptr, nullptr, nullptr, y1, nullptr);

    // 5) proj: [262144 x 256] x [256 x 256]^T -> fp32 NCDHW scatter
    gemm_wmma_kernel<<<dim3(NTOK / 128, 256 / 128), 256, 0, stream>>>(
        y1, wp, prb, 2, nullptr, nullptr, nullptr, nullptr, out);
}
